// TaskAlignedAssigner_65652870086742
// MI455X (gfx1250) — matrix operand order repaired and checked by reading, not compiled, not run on hardware
//
#include <hip/hip_runtime.h>
#include <hip/hip_bf16.h>

// Problem constants (from reference)
constexpr int B = 32, A = 8400, C = 80, M = 32, TOPK = 13;
constexpr float EPSF = 1e-9f;

constexpr size_t NBM  = (size_t)B * M;     // 1024
constexpr size_t NBA  = (size_t)B * A;     // 268800
constexpr size_t NBMA = NBM * (size_t)A;   // 8,601,600

// Workspace layout (bytes)
constexpr size_t OFF_ALIGN = 0;
constexpr size_t OFF_OV    = OFF_ALIGN + NBMA * sizeof(float);
constexpr size_t OFF_MASK  = OFF_OV    + NBMA * sizeof(float);   // u8 [B][A][M]
constexpr size_t OFF_NORM  = OFF_MASK  + NBMA;                   // f32 [B][A]
constexpr size_t OFF_TIDX  = OFF_NORM  + NBA * sizeof(float);    // i32 [B][A]
constexpr size_t OFF_PAM   = OFF_TIDX  + NBA * sizeof(int);      // f32 [B][M]
constexpr size_t OFF_POV   = OFF_PAM   + NBM * sizeof(float);    // f32 [B][M]

typedef __attribute__((ext_vector_type(16))) _Float16 v16h;
typedef __attribute__((ext_vector_type(8)))  float    v8f;

// ---------------- K1: align_metric & overlaps per (b,m) row ----------------
__global__ __launch_bounds__(256) void k1_metrics(
    const float* __restrict__ ps, const float* __restrict__ pb,
    const float* __restrict__ anc, const int* __restrict__ gl,
    const float* __restrict__ gb, const float* __restrict__ mg,
    float* __restrict__ alignw, float* __restrict__ ovlw) {
  const int bm = blockIdx.x;
  const int b  = bm / M;
  const float x1 = gb[bm * 4 + 0], y1 = gb[bm * 4 + 1];
  const float x2 = gb[bm * 4 + 2], y2 = gb[bm * 4 + 3];
  const int   lab   = gl[bm];
  const bool  valid = mg[bm] > 0.f;
  const float ag = (x2 - x1) * (y2 - y1);
  for (int a = threadIdx.x; a < A; a += 256) {
    const float ax = anc[2 * a], ay = anc[2 * a + 1];
    const float* pbb = pb + ((size_t)b * A + a) * 4;
    const float px1 = pbb[0], py1 = pbb[1], px2 = pbb[2], py2 = pbb[3];
    const float dmin = fminf(fminf(ax - x1, ay - y1), fminf(x2 - ax, y2 - ay));
    const bool gm = (dmin > EPSF) && valid;
    const float lx = fmaxf(x1, px1), ly = fmaxf(y1, py1);
    const float rx = fminf(x2, px2), ry = fminf(y2, py2);
    const float w = fmaxf(rx - lx, 0.f), h = fmaxf(ry - ly, 0.f);
    const float inter = w * h;
    const float ap = (px2 - px1) * (py2 - py1);
    const float iou = inter / (ag + ap - inter + EPSF);
    const float ov = gm ? fmaxf(iou, 0.f) : 0.f;
    const float sc = gm ? ps[((size_t)b * A + a) * C + lab] : 0.f;
    const float o2 = ov * ov;
    const size_t o = (size_t)bm * A + a;
    alignw[o] = sc * (o2 * o2 * o2);   // s^1 * iou^6
    ovlw[o]   = ov;
  }
}

// ---------------- K2: top-13 per (b,m) row, scatter mask_pos0 ----------------
__global__ __launch_bounds__(256) void k2_topk(
    const float* __restrict__ alignw, const float* __restrict__ anc,
    const float* __restrict__ gb, const float* __restrict__ mg,
    unsigned char* __restrict__ maskpos) {
  const int bm = blockIdx.x;
  if (mg[bm] <= 0.f) return;  // masked row: mask stays all-zero
  const int b = bm / M, m = bm % M;
  const float* arow = alignw + (size_t)bm * A;
  __shared__ float sv[256];
  __shared__ int   si[256];
  __shared__ int   sel[TOPK];
  const int tid = threadIdx.x;
  for (int it = 0; it < TOPK; ++it) {
    float bv = -1.f; int bi = 0x7fffffff;
    for (int a = tid; a < A; a += 256) {
      bool skip = false;
      for (int j = 0; j < it; ++j) skip |= (sel[j] == a);
      if (skip) continue;
      const float f = arow[a];
      if (f > bv || (f == bv && a < bi)) { bv = f; bi = a; }
    }
    sv[tid] = bv; si[tid] = bi;
    __syncthreads();
    for (int s = 128; s > 0; s >>= 1) {
      if (tid < s) {
        const float v2 = sv[tid + s]; const int i2 = si[tid + s];
        if (v2 > sv[tid] || (v2 == sv[tid] && i2 < si[tid])) { sv[tid] = v2; si[tid] = i2; }
      }
      __syncthreads();
    }
    if (tid == 0) sel[it] = si[0];
    __syncthreads();
  }
  if (tid < TOPK) {
    const int a = sel[tid];
    const float ax = anc[2 * a], ay = anc[2 * a + 1];
    const float x1 = gb[bm * 4 + 0], y1 = gb[bm * 4 + 1];
    const float x2 = gb[bm * 4 + 2], y2 = gb[bm * 4 + 3];
    const float dmin = fminf(fminf(ax - x1, ay - y1), fminf(x2 - ax, y2 - ay));
    if (dmin > EPSF)  // mask_top_k * mask_in_gts * mask_gt
      maskpos[((size_t)b * A + a) * M + m] = 1;
  }
}

// ---------------- K3: resolve multi-assignment, fg & tgt_idx ----------------
__global__ __launch_bounds__(256) void k3_resolve(
    const float* __restrict__ ovlw, unsigned char* __restrict__ maskpos,
    float* __restrict__ out_fg, int* __restrict__ tgtidx) {
  const int t = blockIdx.x * 256 + threadIdx.x;
  if (t >= (int)NBA) return;
  const int b = t / A, a = t % A;
  unsigned char* mp = maskpos + (size_t)t * M;
  unsigned char loc[M];
  int cnt = 0;
#pragma unroll
  for (int m = 0; m < M; ++m) { loc[m] = mp[m]; cnt += loc[m]; }
  int ti = 0;
  if (cnt > 1) {
    float bv = -1.f; int bi = 0;
#pragma unroll
    for (int m = 0; m < M; ++m) {
      const float v = ovlw[((size_t)(b * M + m)) * A + a];
      if (v > bv) { bv = v; bi = m; }     // first-max tie-break (argmax)
    }
#pragma unroll
    for (int m = 0; m < M; ++m) mp[m] = (unsigned char)(m == bi);
    cnt = 1; ti = bi;
  } else {
#pragma unroll
    for (int m = M - 1; m >= 0; --m) if (loc[m]) ti = m;  // first set m
  }
  out_fg[t] = (float)cnt;   // fg_mask > 0 as 0/1
  tgtidx[t] = ti;
}

// ---------------- K4: pos_am / pos_ov per (b,m) ----------------
__global__ __launch_bounds__(256) void k4_pos(
    const float* __restrict__ alignw, const float* __restrict__ ovlw,
    const unsigned char* __restrict__ maskpos,
    float* __restrict__ pos_am, float* __restrict__ pos_ov) {
  const int bm = blockIdx.x;
  const int b = bm / M, m = bm % M;
  __shared__ float sa[256], so[256];
  float ma = 0.f, mo = 0.f;
  for (int a = threadIdx.x; a < A; a += 256) {
    if (maskpos[((size_t)b * A + a) * M + m]) {
      const size_t o = (size_t)bm * A + a;
      ma = fmaxf(ma, alignw[o]);
      mo = fmaxf(mo, ovlw[o]);
    }
  }
  sa[threadIdx.x] = ma; so[threadIdx.x] = mo;
  __syncthreads();
  for (int s = 128; s > 0; s >>= 1) {
    if (threadIdx.x < s) {
      sa[threadIdx.x] = fmaxf(sa[threadIdx.x], sa[threadIdx.x + s]);
      so[threadIdx.x] = fmaxf(so[threadIdx.x], so[threadIdx.x + s]);
    }
    __syncthreads();
  }
  if (threadIdx.x == 0) { pos_am[bm] = sa[0]; pos_ov[bm] = so[0]; }
}

// ---------------- K5: norm per (b,a) + tgt_bboxes gather ----------------
__global__ __launch_bounds__(256) void k5_norm(
    const float* __restrict__ alignw, const unsigned char* __restrict__ maskpos,
    const float* __restrict__ pos_am, const float* __restrict__ pos_ov,
    const float* __restrict__ gb, const int* __restrict__ tgtidx,
    float* __restrict__ normw, float* __restrict__ out_bbox) {
  const int t = blockIdx.x * 256 + threadIdx.x;
  if (t >= (int)NBA) return;
  const int b = t / A, a = t % A;
  const unsigned char* mp = maskpos + (size_t)t * M;
  float nm = 0.f;
#pragma unroll
  for (int m = 0; m < M; ++m) {
    if (mp[m]) {
      const float term = alignw[((size_t)(b * M + m)) * A + a] * pos_ov[b * M + m] /
                         (pos_am[b * M + m] + EPSF);
      nm = fmaxf(nm, term);
    }
  }
  normw[t] = nm;
  const int ti = tgtidx[t];
  const float* g = gb + ((size_t)(b * M + ti)) * 4;
  out_bbox[(size_t)t * 4 + 0] = g[0];
  out_bbox[(size_t)t * 4 + 1] = g[1];
  out_bbox[(size_t)t * 4 + 2] = g[2];
  out_bbox[(size_t)t * 4 + 3] = g[3];
}

// ---------------- K6: target_scores via WMMA: S(16x80) = P^T(16x32) x H(32x80) ----
// One wave per (b, 16-anchor tile). K = M = 32 -> v_wmma_f32_16x16x32_f16, 5 N-tiles.
__global__ __launch_bounds__(32) void k6_scores_wmma(
    const unsigned char* __restrict__ maskpos, const int* __restrict__ gl,
    const float* __restrict__ normw, float* __restrict__ out_scores) {
  constexpr int ATILES = A / 16;  // 525
  const int lane = threadIdx.x;
  const int b    = blockIdx.x / ATILES;
  const int a0   = (blockIdx.x % ATILES) * 16;
  __shared__ int lab[M];
  lab[lane] = gl[b * M + lane];
  __syncthreads();

  const int  n  = lane & 15;
  const bool hi = lane >= 16;

  // A fragment: rows = anchors a0..a0+15, K = m (0..31).
  // ISA 16-bit A 16x32 layout: lanes 0-15 -> K {0..7,16..23}; lanes 16-31 -> K {8..15,24..31}.
  const unsigned char* mrow = maskpos + ((size_t)b * A + a0 + n) * M;
  v16h afrag;
#pragma unroll
  for (int e = 0; e < 16; ++e) {
    const int k = e + ((e >= 8) ? 8 : 0) + (hi ? 8 : 0);
    afrag[e] = mrow[k] ? (_Float16)1.0f : (_Float16)0.0f;
  }

  // norm for the 8 D-rows this lane holds (D: vgpr r -> row r (lanes<16) / r+8)
  float nrm[8];
#pragma unroll
  for (int r = 0; r < 8; ++r)
    nrm[r] = normw[(size_t)b * A + a0 + r + (hi ? 8 : 0)];

#pragma unroll
  for (int t = 0; t < 5; ++t) {   // C = 80 = 5 tiles of 16 classes
    const int c = t * 16 + n;
    // B fragment 32x16: lanes 0-15 -> K {0..7,16..23}; lanes 16-31 -> K {8..15,24..31} (same as A)
    v16h bfrag;
#pragma unroll
    for (int e = 0; e < 16; ++e) {
      const int k = e + ((e >= 8) ? 8 : 0) + (hi ? 8 : 0);
      bfrag[e] = (lab[k] == c) ? (_Float16)1.0f : (_Float16)0.0f;
    }
    v8f acc = {};
    acc = __builtin_amdgcn_wmma_f32_16x16x32_f16(
        /*neg_a=*/false, afrag, /*neg_b=*/false, bfrag,
        /*c_mod=*/(short)0, acc, /*reuse_a=*/false, /*reuse_b=*/false);
#pragma unroll
    for (int r = 0; r < 8; ++r) {
      const int arow = a0 + r + (hi ? 8 : 0);
      out_scores[((size_t)b * A + arow) * C + c] = acc[r] * nrm[r];
    }
  }
}

extern "C" void kernel_launch(void* const* d_in, const int* in_sizes, int n_in,
                              void* d_out, int out_size, void* d_ws, size_t ws_size,
                              hipStream_t stream) {
  const float* ps  = (const float*)d_in[0];  // pd_scores  [B,A,C]
  const float* pb  = (const float*)d_in[1];  // pd_bboxes  [B,A,4]
  const float* anc = (const float*)d_in[2];  // anc_points [A,2]
  const int*   gl  = (const int*)d_in[3];    // gt_labels  [B,M,1]
  const float* gb  = (const float*)d_in[4];  // gt_bboxes  [B,M,4]
  const float* mg  = (const float*)d_in[5];  // mask_gt    [B,M,1]

  char* ws = (char*)d_ws;
  float*         alignw  = (float*)(ws + OFF_ALIGN);
  float*         ovlw    = (float*)(ws + OFF_OV);
  unsigned char* maskpos = (unsigned char*)(ws + OFF_MASK);
  float*         normw   = (float*)(ws + OFF_NORM);
  int*           tgtidx  = (int*)(ws + OFF_TIDX);
  float*         pos_am  = (float*)(ws + OFF_PAM);
  float*         pos_ov  = (float*)(ws + OFF_POV);

  float* out_bbox   = (float*)d_out;                 // [B,A,4]
  float* out_scores = out_bbox + NBA * 4;            // [B,A,C]
  float* out_fg     = out_scores + NBA * (size_t)C;  // [B,A]

  hipMemsetAsync(maskpos, 0, NBMA, stream);
  k1_metrics<<<(int)NBM, 256, 0, stream>>>(ps, pb, anc, gl, gb, mg, alignw, ovlw);
  k2_topk<<<(int)NBM, 256, 0, stream>>>(alignw, anc, gb, mg, maskpos);
  k3_resolve<<<(int)((NBA + 255) / 256), 256, 0, stream>>>(ovlw, maskpos, out_fg, tgtidx);
  k4_pos<<<(int)NBM, 256, 0, stream>>>(alignw, ovlw, maskpos, pos_am, pos_ov);
  k5_norm<<<(int)((NBA + 255) / 256), 256, 0, stream>>>(alignw, maskpos, pos_am, pos_ov,
                                                        gb, tgtidx, normw, out_bbox);
  k6_scores_wmma<<<B * (A / 16), 32, 0, stream>>>(maskpos, gl, normw, out_scores);
}
